// ArcfaceMultiGPU_33732673143454
// MI455X (gfx1250) — matrix operand order repaired and checked by reading, not compiled, not run on hardware
//
#include <hip/hip_runtime.h>

// ---------------------------------------------------------------------------
// ArcFace forward for MI455X (gfx1250): split-f16 WMMA GEMM + fused epilogue.
// Deterministic: no floating-point atomics; per-WG partials reduced in fixed
// order by the loss kernel.
// ---------------------------------------------------------------------------

typedef __attribute__((ext_vector_type(16))) _Float16 v16h;
typedef __attribute__((ext_vector_type(8)))  _Float16 v8h;
typedef __attribute__((ext_vector_type(8)))  float    v8f;
typedef unsigned int       u32;
typedef unsigned long long u64;
typedef __attribute__((ext_vector_type(4))) u32 u32x4;
typedef __attribute__((ext_vector_type(8))) int i32x8;
typedef __attribute__((ext_vector_type(4))) int i32x4;

#define BDIM   512
#define EDIM   512
#define CDIM   100000
#define NT     64            // classes per workgroup
#define NCT    1563          // ceil(100000/64)
#define S_SCALE 64.0f
#define COS_M   0.877582561890372717f
#define SIN_M   0.479425538604203006f
#define MM_C    0.239712769302101503f   // sin(m)*m
#define THRESH  (-0.877582561890372717f) // cos(pi-m)
#define EPSN    1e-6f

// ---- LDS layout (dynamic shared), bytes ----
#define BT_STRIDE   520                       // halves per class row (pad 512->520, 16B aligned, bank stride 4)
#define BT_PLANE    (NT * BT_STRIDE * 2)      // 66560
#define EB_ROWB     80                        // bytes per emb row per chunk (64 data + 16 pad)
#define EB_PLANE    (512 * EB_ROWB)           // 40960
#define OFF_BTH     0
#define OFF_BTL     (OFF_BTH + BT_PLANE)
#define OFF_EB      (OFF_BTL + BT_PLANE)      // [buf][plane] * EB_PLANE (4 planes)
#define OFF_CF      (OFF_EB + 4 * EB_PLANE)   // colfac[64]
#define SMEM_BYTES  (OFF_CF + 64 * 4)         // 297216  (< 320KB WGP LDS)

// ---------------------------------------------------------------------------
// TDM: 2D tensor load (global -> LDS) with LDS row padding.  6-arg builtin
// form (clang-23 / therock); D# bitfields per CDNA5 ISA.
// ---------------------------------------------------------------------------
__device__ __forceinline__ void tdm_load_2d(u32 lds_off, const void* gptr,
                                            u32 tdim0, u32 tdim1,
                                            u32 tile0, u32 tile1,
                                            u64 stride0, u32 flags) {
  u64 ga = (u64)gptr;
  u32x4 g0;
  g0[0] = 1u;                                             // count=1, gather off
  g0[1] = lds_off;                                        // lds_addr
  g0[2] = (u32)ga;                                        // global_addr[31:0]
  g0[3] = ((u32)(ga >> 32) & 0x01FFFFFFu) | (2u << 30);   // addr[56:32] | type=2
  i32x8 g1;
  g1[0] = (int)flags;                                     // mask=0|data_size|pad ctl
  g1[1] = (int)((tdim0 & 0xFFFFu) << 16);                 // tensor_dim0[15:0]
  g1[2] = (int)((tdim0 >> 16) | ((tdim1 & 0xFFFFu) << 16));
  g1[3] = (int)((tdim1 >> 16) | (tile0 << 16));           // tile_dim0
  g1[4] = (int)(tile1 & 0xFFFFu);                         // tile_dim1, tile_dim2=0
  g1[5] = (int)(u32)(stride0 & 0xFFFFFFFFu);              // dim0_stride lo
  g1[6] = (int)(u32)((stride0 >> 32) & 0xFFFFu);          // dim0_stride hi, s1=0
  g1[7] = 0;
  i32x4 z4 = {0, 0, 0, 0};
  i32x8 z8 = {0, 0, 0, 0, 0, 0, 0, 0};
  __builtin_amdgcn_tensor_load_to_lds(g0, g1, z4, z4, z8, 0);
}

// data_size=2B (code 1), pad_enable, pad every 16 DWORDs (code 3) by 4 DWORDs
// (code 3): 64B data rows -> 80B LDS stride (conflict-free b128 frag reads).
#define TDM_FLAGS_EMB ((1u << 16) | (1u << 20) | (3u << 22) | (3u << 25))

// ---------------------------------------------------------------------------
// fragment load: two 16B LDS chunks -> one v16h (8 VGPRs)
// ---------------------------------------------------------------------------
__device__ __forceinline__ v16h ld2(const _Float16* p0, const _Float16* p1) {
  const v8h a = *(const v8h*)__builtin_assume_aligned(p0, 16);
  const v8h b = *(const v8h*)__builtin_assume_aligned(p1, 16);
  v16h r;
#pragma unroll
  for (int i = 0; i < 8; ++i) { r[i] = a[i]; r[i + 8] = b[i]; }
  return r;
}

__device__ __forceinline__ v8f wmma16(v16h a, v16h b, v8f c) {
  return __builtin_amdgcn_wmma_f32_16x16x32_f16(false, a, false, b, (short)0, c,
                                                false, false);
}

// ---------------------------------------------------------------------------
// Kernel 0: split emb f32 -> (hi,lo) f16 planes.
// ---------------------------------------------------------------------------
__global__ void arcface_prep(const float* __restrict__ emb,
                             _Float16* __restrict__ eh, _Float16* __restrict__ el) {
  int i = blockIdx.x * blockDim.x + threadIdx.x;
  if (i < BDIM * EDIM) {
    float x = emb[i];
    _Float16 h = (_Float16)x;
    eh[i] = h;
    el[i] = (_Float16)(x - (float)h);
  }
}

// ---------------------------------------------------------------------------
// Kernel 1: per-WG: stage kernel-tile (transposed, split f16) + column norms,
// split-f16 WMMA GEMM over all 512 rows, fused arcface epilogue + per-WG
// partial sum-exp (deterministic, no atomics).  emb chunks streamed via TDM,
// double buffered.
// ---------------------------------------------------------------------------
__global__ void __launch_bounds__(256)
arcface_main(const float* __restrict__ kern, const int* __restrict__ label,
             const _Float16* __restrict__ embH, const _Float16* __restrict__ embL,
             float* __restrict__ partial, float* __restrict__ dout) {
  extern __shared__ char smem[];
  _Float16* BtH = (_Float16*)(smem + OFF_BTH);
  _Float16* BtL = (_Float16*)(smem + OFF_BTL);
  float* colfac = (float*)(smem + OFF_CF);

  const int tid = threadIdx.x;
  const int wv = tid >> 5;          // wave 0..7  (wave32)
  const int ln = tid & 31;
  const int half = ln >> 4;         // 0: lanes 0-15, 1: lanes 16-31
  const int lr = ln & 15;
  const int n0 = blockIdx.x * NT;   // first class of this tile

  // wave0 kicks off TDM prefetch of emb chunk 0 (both planes) into buffer 0.
  if (wv == 0) {
    tdm_load_2d((u32)(u64)(smem + OFF_EB + 0 * EB_PLANE), embH, 512, 512, 32, 512,
                512, TDM_FLAGS_EMB);
    tdm_load_2d((u32)(u64)(smem + OFF_EB + 1 * EB_PLANE), embL, 512, 512, 32, 512,
                512, TDM_FLAGS_EMB);
  }

  // ---- Phase A: stage kernel tile transposed as hi/lo f16.
  for (int it = tid; it < 512 * (NT / 4); it += 256) {
    const int k = it >> 4;
    const int nq = (it & 15) * 4;
    const int col = n0 + nq;
    float4 v = make_float4(0.f, 0.f, 0.f, 0.f);
    if (col < CDIM) v = *(const float4*)(kern + (u64)k * CDIM + col);
    const float xs[4] = {v.x, v.y, v.z, v.w};
#pragma unroll
    for (int j = 0; j < 4; ++j) {
      const float x = xs[j];
      const _Float16 h = (_Float16)x;
      BtH[(nq + j) * BT_STRIDE + k] = h;
      BtL[(nq + j) * BT_STRIDE + k] = (_Float16)(x - (float)h);
    }
  }
  __syncthreads();

  // column norms from the staged tile (hi+lo reconstruct, err ~2^-22)
  if (tid < NT) {
    float s = 0.f;
    for (int k = 0; k < 512; k += 8) {
      const v8h h = *(const v8h*)__builtin_assume_aligned(&BtH[tid * BT_STRIDE + k], 16);
      const v8h l = *(const v8h*)__builtin_assume_aligned(&BtL[tid * BT_STRIDE + k], 16);
#pragma unroll
      for (int j = 0; j < 8; ++j) {
        const float x = (float)h[j] + (float)l[j];
        s += x * x;
      }
    }
    colfac[tid] = (n0 + tid < CDIM) ? (1.f / (sqrtf(s) + EPSN)) : 0.f;
  }

  // per-lane labels for this wave's 4 row-tiles
  int labv[4];
#pragma unroll
  for (int mt = 0; mt < 4; ++mt) labv[mt] = label[wv * 64 + mt * 16 + lr];

  // ---- K loop: 16 chunks of K=32, double-buffered TDM emb staging ----
  v8f acc[4][4];
#pragma unroll
  for (int ct = 0; ct < 4; ++ct)
#pragma unroll
    for (int mt = 0; mt < 4; ++mt) acc[ct][mt] = (v8f){0, 0, 0, 0, 0, 0, 0, 0};

#pragma unroll 1
  for (int kc = 0; kc < 16; ++kc) {
    const int buf = kc & 1;
    if (wv == 0) {
      if (kc + 1 < 16) {
        const int nb = buf ^ 1;
        tdm_load_2d((u32)(u64)(smem + OFF_EB + (nb * 2 + 0) * EB_PLANE),
                    embH + (kc + 1) * 32, 512, 512, 32, 512, 512, TDM_FLAGS_EMB);
        tdm_load_2d((u32)(u64)(smem + OFF_EB + (nb * 2 + 1) * EB_PLANE),
                    embL + (kc + 1) * 32, 512, 512, 32, 512, 512, TDM_FLAGS_EMB);
        __builtin_amdgcn_s_wait_tensorcnt(2);   // chunk kc complete
      } else {
        __builtin_amdgcn_s_wait_tensorcnt(0);
      }
    }
    __syncthreads();   // chunk kc visible to all waves

    const _Float16* ebH = (const _Float16*)(smem + OFF_EB + (buf * 2 + 0) * EB_PLANE);
    const _Float16* ebL = (const _Float16*)(smem + OFF_EB + (buf * 2 + 1) * EB_PLANE);

    // B fragments (emb rows, 16 consecutive K per lane), hi & lo planes
    v16h bh[4], bl[4];
#pragma unroll
    for (int mt = 0; mt < 4; ++mt) {
      const int m = wv * 64 + mt * 16 + lr;
      const _Float16* r0 = ebH + m * (EB_ROWB / 2) + half * 16;
      const _Float16* r1 = ebL + m * (EB_ROWB / 2) + half * 16;
      bh[mt] = ld2(r0, r0 + 8);
      bl[mt] = ld2(r1, r1 + 8);
    }

#pragma unroll
    for (int ct = 0; ct < 4; ++ct) {
      // A fragments (kernel^T classes; K chunks {0-15}/{16-31})
      const int aoff = (ct * 16 + lr) * BT_STRIDE + kc * 32 + half * 16;
      const v16h ah = ld2(BtH + aoff, BtH + aoff + 8);
      const v16h al = ld2(BtL + aoff, BtL + aoff + 8);
#pragma unroll
      for (int mt = 0; mt < 4; ++mt) {
        acc[ct][mt] = wmma16(ah, bh[mt], acc[ct][mt]);
        acc[ct][mt] = wmma16(ah, bl[mt], acc[ct][mt]);
        acc[ct][mt] = wmma16(al, bh[mt], acc[ct][mt]);
      }
    }
    __syncthreads();   // buffer consumed before TDM overwrites it
  }

  // ---- Epilogue: normalize, clip, arcface margin, stores, partial sum-exp ---
  float* outp = dout + 1;
  float* cosp = dout + 1 + (u64)BDIM * CDIM;
#pragma unroll
  for (int mt = 0; mt < 4; ++mt) {
    const int mrow = wv * 64 + mt * 16 + lr;
    const int lab = labv[mt];
    float se = 0.f;
#pragma unroll
    for (int ct = 0; ct < 4; ++ct) {
      const int cloc = ct * 16 + half * 8;     // local class base for this lane
      const v8f a = acc[ct][mt];
      const u64 obase = (u64)mrow * CDIM + (u64)(n0 + cloc);
#pragma unroll
      for (int r = 0; r < 8; ++r) {
        const int cls = n0 + cloc + r;
        if (cls < CDIM) {
          float c = a[r] * colfac[cloc + r];
          c = fminf(1.f, fmaxf(-1.f, c));
          float o = c;
          if (cls == lab) {
            const float s = sqrtf(fmaxf(0.f, 1.f - c * c));
            const float cm = c * COS_M - s * SIN_M;
            o = (c - THRESH > 0.f) ? cm : (c - MM_C);
          }
          o *= S_SCALE;
          outp[obase + r] = o;
          cosp[obase + r] = c;
          se += __expf(o);
        }
      }
    }
    // lanes l and l+16 hold the same batch row: combine once, single writer.
    const float tot = se + __shfl_down(se, 16);
    if (half == 0) partial[(u64)blockIdx.x * BDIM + mrow] = tot;
  }
}

// ---------------------------------------------------------------------------
// Kernel 2: loss = mean_i( log(sum_wg partial[wg][i]) - output[i, label_i] )
// Fixed reduction order -> bitwise deterministic across replays.
// ---------------------------------------------------------------------------
__global__ void arcface_loss(const float* __restrict__ partial,
                             const int* __restrict__ label,
                             float* __restrict__ dout) {
  __shared__ float red[BDIM];
  const int i = threadIdx.x;
  float s = 0.f;
  for (int wg = 0; wg < NCT; ++wg) s += partial[(u64)wg * BDIM + i];
  const float olab = dout[1 + (u64)i * CDIM + label[i]];
  red[i] = logf(s) - olab;
  __syncthreads();
  for (int st = BDIM / 2; st > 0; st >>= 1) {
    if (i < st) red[i] += red[i + st];
    __syncthreads();
  }
  if (i == 0) dout[0] = red[0] * (1.f / (float)BDIM);
}

// ---------------------------------------------------------------------------
extern "C" void kernel_launch(void* const* d_in, const int* in_sizes, int n_in,
                              void* d_out, int out_size, void* d_ws, size_t ws_size,
                              hipStream_t stream) {
  const float* emb  = (const float*)d_in[0];   // [512,512] f32 (unit rows)
  const float* kern = (const float*)d_in[1];   // [512,100000] f32
  const int*   lab  = (const int*)d_in[2];     // [512] i32
  float* out = (float*)d_out;                  // [1 + 2*512*100000] f32

  _Float16* eh = (_Float16*)d_ws;                                  // 512 KB
  _Float16* el = eh + BDIM * EDIM;                                 // 512 KB
  float* partial = (float*)((char*)d_ws + (size_t)BDIM * EDIM * 4); // 1563*512*4 B

  arcface_prep<<<(BDIM * EDIM) / 256, 256, 0, stream>>>(emb, eh, el);
  arcface_main<<<NCT, 256, SMEM_BYTES, stream>>>(kern, lab, eh, el, partial, out);
  arcface_loss<<<1, BDIM, 0, stream>>>(partial, lab, out);
}
